// RASG_3298534884252
// MI455X (gfx1250) — matrix operand order repaired and checked by reading, not compiled, not run on hardware
//
#include <hip/hip_runtime.h>
#include <hip/hip_fp16.h>
#include <stdint.h>

#define N_NODES 50000
#define N_EDGES 1600000
#define BGRAPHS 512
#define HID 128
#define IN_D 64
#define EPS 1e-5f

typedef __attribute__((ext_vector_type(16))) _Float16 v16h;
typedef __attribute__((ext_vector_type(8)))  float    v8f;

__device__ __forceinline__ float leaky(float x) { return x > 0.f ? x : 0.1f * x; }

__device__ __forceinline__ float wave_sum(float v) {
#pragma unroll
    for (int m = 16; m >= 1; m >>= 1) v += __shfl_xor(v, m, 32);
    return v;
}

__device__ __forceinline__ void atomicMaxF(float* addr, float v) {
    if (v >= 0.f) atomicMax((int*)addr, __float_as_int(v));
    else          atomicMin((unsigned int*)addr, __float_as_uint(v));
}

// ---------------------------------------------------------------------------
// Node embedding: x = MLP(labels), rel_feat = LN(leaky(table[rel]@re_w+b))
// One wave (32 lanes) per node; lane l owns output feature l of each 32-vector.
// ---------------------------------------------------------------------------
__global__ void embed_kernel(const int* __restrict__ xlab, const int* __restrict__ batch,
                             const int* __restrict__ rel_ids,
                             const float* __restrict__ ne_w1, const float* __restrict__ ne_b1,
                             const float* __restrict__ ne_w2, const float* __restrict__ ne_b2,
                             const float* __restrict__ re_table, const float* __restrict__ re_w,
                             const float* __restrict__ re_b, const float* __restrict__ re_ln_g,
                             const float* __restrict__ re_ln_b, float* __restrict__ h0) {
    int node = blockIdx.x * (blockDim.x >> 5) + (threadIdx.x >> 5);
    int lane = threadIdx.x & 31;
    if (node >= N_NODES) return;
    float n0 = (float)xlab[node * 2 + 0] * (1.0f / 10.00001f);
    float n1 = (float)xlab[node * 2 + 1] * (1.0f / 10.00001f);
    float x = ne_b2[lane];
#pragma unroll 4
    for (int j = 0; j < 64; ++j) {
        float t = leaky(n0 * ne_w1[j] + n1 * ne_w1[64 + j] + ne_b1[j]);
        x += t * ne_w2[j * 32 + lane];
    }
    int rel = rel_ids[batch[node]];
    const float* rv = re_table + (size_t)rel * 32;
    float u = re_b[lane];
#pragma unroll 4
    for (int j = 0; j < 32; ++j) u += rv[j] * re_w[j * 32 + lane];
    u = leaky(u);
    float mean = wave_sum(u) * (1.f / 32.f);
    float d = u - mean;
    float var = wave_sum(d * d) * (1.f / 32.f);
    float y = d * rsqrtf(var + EPS) * re_ln_g[lane] + re_ln_b[lane];
    h0[(size_t)node * IN_D + lane]      = x;
    h0[(size_t)node * IN_D + 32 + lane] = y;
}

// ---------------------------------------------------------------------------
// Edge aggregation: aggr[dst] += h[src].  One wave per edge; lanes stride the
// feature dim for coalesced 128B reads + global_atomic_add_f32 (L2-resident).
// ---------------------------------------------------------------------------
__global__ void scatter_kernel(const float* __restrict__ H, const int* __restrict__ srcv,
                               const int* __restrict__ dstv, float* __restrict__ Aggr, int K) {
    int e = blockIdx.x * (blockDim.x >> 5) + (threadIdx.x >> 5);
    int lane = threadIdx.x & 31;
    if (e >= N_EDGES) return;
    int s = srcv[e], d = dstv[e];
    const float* hp = H + (size_t)s * K;
    float* ap = Aggr + (size_t)d * K;
    for (int j = lane; j < K; j += 32) atomicAdd(ap + j, hp[j]);
}

// ---------------------------------------------------------------------------
// WMMA fragment helpers (documented gfx1250 16-bit A(16x32)/B(32x16) layouts).
// A: loaded per-GEMM from f32 activations (float2 -> cvt_pk_f16).
// B: weights are pre-packed once into per-lane v16h fragments (f16), so the
//    GEMM inner loop reads each fragment as a single 32B/lane vector load.
// ---------------------------------------------------------------------------
__device__ __forceinline__ v16h load_a_frag(const float* __restrict__ A, int lda,
                                            int row_base, int kb) {
    int lane = threadIdx.x & 31;
    int half = lane >> 4;
    const float* rp = A + (size_t)(row_base + (lane & 15)) * lda + kb;
    v16h a;
#pragma unroll
    for (int i = 0; i < 8; ++i) {
        int k0 = ((i < 4) ? 2 * i : 16 + 2 * (i - 4)) + (half ? 8 : 0);
        float2 f = *(const float2*)(rp + k0);
        a[2 * i]     = (_Float16)f.x;
        a[2 * i + 1] = (_Float16)f.y;
    }
    return a;
}

__device__ __forceinline__ v16h load_b_frag_f32(const float* __restrict__ Bm, int ldb,
                                                int kb, int col_base) {
    int lane = threadIdx.x & 31;
    int half = lane >> 4;
    const float* cp = Bm + (size_t)kb * ldb + col_base + (lane & 15);
    v16h b;
#pragma unroll
    for (int i = 0; i < 8; ++i) {
        int k0 = ((i < 4) ? 2 * i : 16 + 2 * (i - 4)) + (half ? 8 : 0);
        b[2 * i]     = (_Float16)cp[(size_t)k0 * ldb];
        b[2 * i + 1] = (_Float16)cp[(size_t)(k0 + 1) * ldb];
    }
    return b;
}

// Pack a (K x 128) f32 weight matrix into f16 B-fragments.
// Fragment slot layout: frag[(kb/32)*8 + tile][lane][16 halves].
__global__ void pack_w_kernel(const float* __restrict__ W, int K, _Float16* __restrict__ out) {
    int wave = blockIdx.x * (blockDim.x >> 5) + (threadIdx.x >> 5);
    int nfrag = (K >> 5) * 8;
    if (wave >= nfrag) return;
    int kb = (wave >> 3) * 32;
    int t  = wave & 7;
    v16h b = load_b_frag_f32(W, HID, kb, t * 16);
    int lane = threadIdx.x & 31;
    *(v16h*)(out + ((size_t)wave * 32 + lane) * 16) = b;
}

__device__ __forceinline__ v16h load_b_packed(const _Float16* __restrict__ P, int kb, int t) {
    int lane = threadIdx.x & 31;
    size_t slot = ((size_t)(kb >> 5) * 8 + t) * 32 + lane;
    return *(const v16h*)(P + slot * 16);
}

// ---------------------------------------------------------------------------
// Fused dual GEMM: Out = Aggr@Wrel + H@Wnode + bnode.   One wave computes a
// 16-row x 128-col strip with 8 v_wmma_f32_16x16x32_f16 accumulators.
// N_NODES is a multiple of 16, so no row tail handling is needed.
// ---------------------------------------------------------------------------
__global__ void gemm_dual_kernel(const float* __restrict__ Aggr, const float* __restrict__ H,
                                 const _Float16* __restrict__ PWrel,
                                 const _Float16* __restrict__ PWnode,
                                 const float* __restrict__ bnode, float* __restrict__ Out, int K) {
    int tile = blockIdx.x * (blockDim.x >> 5) + (threadIdx.x >> 5);
    int row_base = tile * 16;
    if (row_base >= N_NODES) return;
    int lane = threadIdx.x & 31;
    v8f c[8] = {};
    for (int kb = 0; kb < K; kb += 32) {
        v16h a = load_a_frag(Aggr, K, row_base, kb);
#pragma unroll
        for (int t = 0; t < 8; ++t) {
            v16h b = load_b_packed(PWrel, kb, t);
            c[t] = __builtin_amdgcn_wmma_f32_16x16x32_f16(false, a, false, b,
                                                          (short)0, c[t], false, false);
        }
    }
    for (int kb = 0; kb < K; kb += 32) {
        v16h a = load_a_frag(H, K, row_base, kb);
#pragma unroll
        for (int t = 0; t < 8; ++t) {
            v16h b = load_b_packed(PWnode, kb, t);
            c[t] = __builtin_amdgcn_wmma_f32_16x16x32_f16(false, a, false, b,
                                                          (short)0, c[t], false, false);
        }
    }
    int half = lane >> 4;
    int cix  = lane & 15;
#pragma unroll
    for (int t = 0; t < 8; ++t) {
        int col = t * 16 + cix;
        float bb = bnode[col];
#pragma unroll
        for (int r = 0; r < 8; ++r) {
            int row = row_base + r + half * 8;
            Out[(size_t)row * HID + col] = c[t][r] + bb;
        }
    }
}

// ---------------------------------------------------------------------------
// Fused BatchNorm(eval)+leaky+LayerNorm(+optional leaky), in place.
// One wave per node, 4 features per lane.
// ---------------------------------------------------------------------------
__global__ void bnln_kernel(float* __restrict__ X, const float* __restrict__ bng,
                            const float* __restrict__ bnb, const float* __restrict__ lng,
                            const float* __restrict__ lnb, int do_leaky) {
    int node = blockIdx.x * (blockDim.x >> 5) + (threadIdx.x >> 5);
    int lane = threadIdx.x & 31;
    if (node >= N_NODES) return;
    float* row = X + (size_t)node * HID;
    const float inv = rsqrtf(1.0f + EPS);
    float v[4];
    float s = 0.f;
#pragma unroll
    for (int m = 0; m < 4; ++m) {
        int c = lane + 32 * m;
        float t = row[c] * inv * bng[c] + bnb[c];
        t = leaky(t);
        v[m] = t;
        s += t;
    }
    float mean = wave_sum(s) * (1.f / HID);
    float ss = 0.f;
#pragma unroll
    for (int m = 0; m < 4; ++m) { float d = v[m] - mean; ss += d * d; }
    float var = wave_sum(ss) * (1.f / HID);
    float rs = rsqrtf(var + EPS);
#pragma unroll
    for (int m = 0; m < 4; ++m) {
        int c = lane + 32 * m;
        float y = (v[m] - mean) * rs * lng[c] + lnb[c];
        if (do_leaky) y = leaky(y);
        row[c] = y;
    }
}

__global__ void att_init_kernel(float* mbuf, float* sbuf, float* pooled) {
    int i = blockIdx.x * blockDim.x + threadIdx.x;
    if (i < BGRAPHS * 4) { mbuf[i] = -__builtin_huge_valf(); sbuf[i] = 0.f; }
    if (i < BGRAPHS * 256) pooled[i] = 0.f;
}

// Pass 1: per-node attention logits + segment max. One wave per node;
// lane l owns key columns {l, l+32, ..., l+224}.
__global__ void att_score_kernel(const float* __restrict__ H, const int* __restrict__ batch,
                                 const float* __restrict__ att_proj,
                                 const float* __restrict__ att_query,
                                 float* __restrict__ att, float* __restrict__ mbuf) {
    int node = blockIdx.x * (blockDim.x >> 5) + (threadIdx.x >> 5);
    int lane = threadIdx.x & 31;
    if (node >= N_NODES) return;
    const float* hr = H + (size_t)node * HID;
    float key[8] = {0.f, 0.f, 0.f, 0.f, 0.f, 0.f, 0.f, 0.f};
    for (int i = 0; i < HID; ++i) {
        float hv = hr[i];
        const float* pr = att_proj + (size_t)i * 256 + lane;
#pragma unroll
        for (int m = 0; m < 8; ++m) key[m] += hv * pr[32 * m];
    }
    float p0 = 0.f, p1 = 0.f, p2 = 0.f, p3 = 0.f;
#pragma unroll
    for (int m = 0; m < 8; ++m) {
        int d = lane + 32 * (m & 1);
        float term = key[m] * att_query[(m >> 1) * 64 + d];
        if ((m >> 1) == 0) p0 += term;
        else if ((m >> 1) == 1) p1 += term;
        else if ((m >> 1) == 2) p2 += term;
        else p3 += term;
    }
    p0 = wave_sum(p0); p1 = wave_sum(p1); p2 = wave_sum(p2); p3 = wave_sum(p3);
    if (lane < 4) {
        float pv = (lane == 0) ? p0 : (lane == 1) ? p1 : (lane == 2) ? p2 : p3;
        att[(size_t)node * 4 + lane] = pv;
        int b = batch[node];
        atomicMaxF(&mbuf[b * 4 + lane], pv);
    }
}

// Pass 2: a = exp(att - m); accumulate s and un-normalized pooled = sum(a*key).
__global__ void att_pool_kernel(const float* __restrict__ H, const int* __restrict__ batch,
                                const float* __restrict__ att_proj,
                                const float* __restrict__ att, const float* __restrict__ mbuf,
                                float* __restrict__ sbuf, float* __restrict__ pooled) {
    int node = blockIdx.x * (blockDim.x >> 5) + (threadIdx.x >> 5);
    int lane = threadIdx.x & 31;
    if (node >= N_NODES) return;
    const float* hr = H + (size_t)node * HID;
    float key[8] = {0.f, 0.f, 0.f, 0.f, 0.f, 0.f, 0.f, 0.f};
    for (int i = 0; i < HID; ++i) {
        float hv = hr[i];
        const float* pr = att_proj + (size_t)i * 256 + lane;
#pragma unroll
        for (int m = 0; m < 8; ++m) key[m] += hv * pr[32 * m];
    }
    int b = batch[node];
    float a0 = expf(att[(size_t)node * 4 + 0] - mbuf[b * 4 + 0]);
    float a1 = expf(att[(size_t)node * 4 + 1] - mbuf[b * 4 + 1]);
    float a2 = expf(att[(size_t)node * 4 + 2] - mbuf[b * 4 + 2]);
    float a3 = expf(att[(size_t)node * 4 + 3] - mbuf[b * 4 + 3]);
    if (lane < 4) {
        float av = (lane == 0) ? a0 : (lane == 1) ? a1 : (lane == 2) ? a2 : a3;
        atomicAdd(&sbuf[b * 4 + lane], av);
    }
#pragma unroll
    for (int m = 0; m < 8; ++m) {
        float av = ((m >> 1) == 0) ? a0 : ((m >> 1) == 1) ? a1 : ((m >> 1) == 2) ? a2 : a3;
        atomicAdd(&pooled[(size_t)b * 256 + lane + 32 * m], av * key[m]);
    }
}

// Final: normalize pooled, LayerNorm -> z_graph; gather head/tail reprs;
// scorer MLP 512->128->64->1.  One wave per graph.
__global__ void final_kernel(const float* __restrict__ pooled, const float* __restrict__ sbuf,
                             const float* __restrict__ H, const int* __restrict__ head_idx,
                             const int* __restrict__ tail_idx,
                             const float* __restrict__ att_ln_g, const float* __restrict__ att_ln_b,
                             const float* __restrict__ sc_w1, const float* __restrict__ sc_b1,
                             const float* __restrict__ sc_w2, const float* __restrict__ sc_b2,
                             const float* __restrict__ sc_w3, const float* __restrict__ sc_b3,
                             float* __restrict__ out) {
    __shared__ float feat[512];
    __shared__ float z1[128];
    __shared__ float z2[64];
    int b = blockIdx.x;
    int lane = threadIdx.x;
    float v[8];
    float s = 0.f;
#pragma unroll
    for (int m = 0; m < 8; ++m) {
        int c = lane + 32 * m;
        float t = pooled[(size_t)b * 256 + c] / sbuf[b * 4 + (c >> 6)];
        v[m] = t;
        s += t;
    }
    float mean = wave_sum(s) * (1.f / 256.f);
    float ss = 0.f;
#pragma unroll
    for (int m = 0; m < 8; ++m) { float d = v[m] - mean; ss += d * d; }
    float var = wave_sum(ss) * (1.f / 256.f);
    float rs = rsqrtf(var + EPS);
#pragma unroll
    for (int m = 0; m < 8; ++m) {
        int c = lane + 32 * m;
        feat[c] = (v[m] - mean) * rs * att_ln_g[c] + att_ln_b[c];
    }
    int hi = head_idx[b], ti = tail_idx[b];
    int hic = hi > 0 ? hi : 0, tic = ti > 0 ? ti : 0;
#pragma unroll
    for (int m = 0; m < 4; ++m) {
        int j = lane + 32 * m;
        feat[256 + j] = (hi >= 0) ? H[(size_t)hic * HID + j] : 0.f;
        feat[384 + j] = (ti >= 0) ? H[(size_t)tic * HID + j] : 0.f;
    }
    __syncthreads();
#pragma unroll
    for (int m = 0; m < 4; ++m) {
        int j = lane + 32 * m;
        float acc = sc_b1[j];
        for (int i = 0; i < 512; ++i) acc += feat[i] * sc_w1[(size_t)i * 128 + j];
        z1[j] = leaky(acc);
    }
    __syncthreads();
#pragma unroll
    for (int m = 0; m < 2; ++m) {
        int j = lane + 32 * m;
        float acc = sc_b2[j];
        for (int i = 0; i < 128; ++i) acc += z1[i] * sc_w2[(size_t)i * 64 + j];
        z2[j] = leaky(acc);
    }
    __syncthreads();
    float acc = z2[lane] * sc_w3[lane] + z2[lane + 32] * sc_w3[lane + 32];
    acc = wave_sum(acc);
    if (lane == 0) out[b] = acc + sc_b3[0];
}

extern "C" void kernel_launch(void* const* d_in, const int* in_sizes, int n_in,
                              void* d_out, int out_size, void* d_ws, size_t ws_size,
                              hipStream_t stream) {
    const int* xlab     = (const int*)d_in[0];
    const int* eidx     = (const int*)d_in[1];
    const int* batch    = (const int*)d_in[2];
    const int* head_idx = (const int*)d_in[3];
    const int* tail_idx = (const int*)d_in[4];
    const int* rel_ids  = (const int*)d_in[5];
    const float* ne_w1  = (const float*)d_in[6];
    const float* ne_b1  = (const float*)d_in[7];
    const float* ne_w2  = (const float*)d_in[8];
    const float* ne_b2  = (const float*)d_in[9];
    const float* re_table = (const float*)d_in[10];
    const float* re_w   = (const float*)d_in[11];
    const float* re_b   = (const float*)d_in[12];
    const float* re_ln_g = (const float*)d_in[13];
    const float* re_ln_b = (const float*)d_in[14];
    const float* att_proj  = (const float*)d_in[15];
    const float* att_query = (const float*)d_in[16];
    const float* att_ln_g  = (const float*)d_in[17];
    const float* att_ln_b  = (const float*)d_in[18];
    const float* sc_w1 = (const float*)d_in[19];
    const float* sc_b1 = (const float*)d_in[20];
    const float* sc_w2 = (const float*)d_in[21];
    const float* sc_b2 = (const float*)d_in[22];
    const float* sc_w3 = (const float*)d_in[23];
    const float* sc_b3 = (const float*)d_in[24];
    const float *wrel[3], *wnode[3], *bnode[3], *bng[3], *bnb[3], *lng[3], *lnb[3];
    for (int i = 0; i < 3; ++i) {
        int base = 25 + i * 7;
        wrel[i]  = (const float*)d_in[base + 0];
        wnode[i] = (const float*)d_in[base + 1];
        bnode[i] = (const float*)d_in[base + 2];
        bng[i]   = (const float*)d_in[base + 3];
        bnb[i]   = (const float*)d_in[base + 4];
        lng[i]   = (const float*)d_in[base + 5];
        lnb[i]   = (const float*)d_in[base + 6];
    }
    const int* srcv = eidx;
    const int* dstv = eidx + N_EDGES;

    float* ws = (float*)d_ws;
    size_t o = 0;
    float* hbuf0 = ws + o; o += (size_t)N_NODES * HID;
    float* hbuf1 = ws + o; o += (size_t)N_NODES * HID;
    float* aggr  = ws + o; o += (size_t)N_NODES * HID;
    float* attb  = ws + o; o += (size_t)N_NODES * 4;
    float* mbuf  = ws + o; o += BGRAPHS * 4;
    float* sbuf  = ws + o; o += BGRAPHS * 4;
    float* pooled = ws + o; o += BGRAPHS * 256;
    // packed f16 weight fragments: (K/32)*8 frags * 32 lanes * 16 halves
    _Float16* pw_rel[3];
    _Float16* pw_node[3];
    const int kdim[3] = {64, 128, 128};
    for (int i = 0; i < 3; ++i) {
        size_t halves = (size_t)(kdim[i] >> 5) * 8 * 32 * 16;   // multiple of 8 floats
        pw_rel[i]  = (_Float16*)(ws + o); o += halves / 2;
        pw_node[i] = (_Float16*)(ws + o); o += halves / 2;
    }
    (void)ws_size; (void)in_sizes; (void)n_in; (void)out_size;

    dim3 blk(256);
    int gridNodes = (N_NODES + 7) / 8;
    int gridEdges = (N_EDGES + 7) / 8;
    int gridTiles = (N_NODES / 16 + 7) / 8;

    // Pre-pack weight matrices into f16 WMMA B-fragments (tiny, once per call).
    for (int i = 0; i < 3; ++i) {
        int nfrag = (kdim[i] >> 5) * 8;
        int g = (nfrag + 7) / 8;
        pack_w_kernel<<<g, blk, 0, stream>>>(wrel[i],  kdim[i], pw_rel[i]);
        pack_w_kernel<<<g, blk, 0, stream>>>(wnode[i], kdim[i], pw_node[i]);
    }

    // h0 (N x 64): [node_emb | rel_emb]
    embed_kernel<<<gridNodes, blk, 0, stream>>>(xlab, batch, rel_ids, ne_w1, ne_b1, ne_w2, ne_b2,
                                                re_table, re_w, re_b, re_ln_g, re_ln_b, hbuf0);
    // Layer 0 (K = 64)
    hipMemsetAsync(aggr, 0, (size_t)N_NODES * 64 * sizeof(float), stream);
    scatter_kernel<<<gridEdges, blk, 0, stream>>>(hbuf0, srcv, dstv, aggr, 64);
    gemm_dual_kernel<<<gridTiles, blk, 0, stream>>>(aggr, hbuf0, pw_rel[0], pw_node[0], bnode[0], hbuf1, 64);
    bnln_kernel<<<gridNodes, blk, 0, stream>>>(hbuf1, bng[0], bnb[0], lng[0], lnb[0], 1);
    // Layer 1 (K = 128)
    hipMemsetAsync(aggr, 0, (size_t)N_NODES * 128 * sizeof(float), stream);
    scatter_kernel<<<gridEdges, blk, 0, stream>>>(hbuf1, srcv, dstv, aggr, 128);
    gemm_dual_kernel<<<gridTiles, blk, 0, stream>>>(aggr, hbuf1, pw_rel[1], pw_node[1], bnode[1], hbuf0, 128);
    bnln_kernel<<<gridNodes, blk, 0, stream>>>(hbuf0, bng[1], bnb[1], lng[1], lnb[1], 1);
    // Layer 2 (K = 128, no trailing leaky)
    hipMemsetAsync(aggr, 0, (size_t)N_NODES * 128 * sizeof(float), stream);
    scatter_kernel<<<gridEdges, blk, 0, stream>>>(hbuf0, srcv, dstv, aggr, 128);
    gemm_dual_kernel<<<gridTiles, blk, 0, stream>>>(aggr, hbuf0, pw_rel[2], pw_node[2], bnode[2], hbuf1, 128);
    bnln_kernel<<<gridNodes, blk, 0, stream>>>(hbuf1, bng[2], bnb[2], lng[2], lnb[2], 0);
    // Attention pooling
    att_init_kernel<<<(BGRAPHS * 256 + 255) / 256, 256, 0, stream>>>(mbuf, sbuf, pooled);
    att_score_kernel<<<gridNodes, blk, 0, stream>>>(hbuf1, batch, att_proj, att_query, attb, mbuf);
    att_pool_kernel<<<gridNodes, blk, 0, stream>>>(hbuf1, batch, att_proj, attb, mbuf, sbuf, pooled);
    // Scorer
    final_kernel<<<BGRAPHS, 32, 0, stream>>>(pooled, sbuf, hbuf1, head_idx, tail_idx,
                                             att_ln_g, att_ln_b, sc_w1, sc_b1, sc_w2, sc_b2,
                                             sc_w3, sc_b3, (float*)d_out);
}
